// LSTMRegressor_53781580480518
// MI455X (gfx1250) — matrix operand order repaired and checked by reading, not compiled, not run on hardware
//
#include <hip/hip_runtime.h>
#include <stdint.h>

// ---------------------------------------------------------------------------
// Problem constants (from reference): B=256, T=512, D=64, H=512, FC=512, OUT=8
// ---------------------------------------------------------------------------
#define Bn   256
#define Tn   512
#define Dn   64
#define Hn   512
#define FCn  512
#define OUTn 8

typedef __attribute__((ext_vector_type(16))) __bf16 v16bf;
typedef __attribute__((ext_vector_type(8)))  float  v8f;

union FragU {
    v16bf  v;
    float4 f[2];
};

#if defined(__has_builtin)
#if __has_builtin(__builtin_amdgcn_global_load_async_to_lds_b128)
#define HAVE_ASYNC_LDS 1
#endif
#endif

#ifdef HAVE_ASYNC_LDS
typedef int v4i __attribute__((vector_size(4 * sizeof(int))));
typedef __attribute__((address_space(1))) v4i* gas1_v4i;
typedef __attribute__((address_space(3))) v4i* as3_v4i;
#endif

__device__ __forceinline__ uint16_t f2bf(float x) {
    union { float f; uint32_t u; } c;
    c.f = x;
    uint32_t r = c.u + 0x7FFFu + ((c.u >> 16) & 1u);   // round-to-nearest-even
    return (uint16_t)(r >> 16);
}

__device__ __forceinline__ float sigf(float x) {
    return 1.0f / (1.0f + __expf(-x));
}

// ---------------------------------------------------------------------------
// fp32 -> bf16 conversion
// ---------------------------------------------------------------------------
__global__ void k_f32_to_bf16(const float* __restrict__ in,
                              uint16_t* __restrict__ out, int n) {
    int i = blockIdx.x * blockDim.x + threadIdx.x;
    if (i < n) out[i] = f2bf(in[i]);
}

// Pack [W_ih | W_hh] row-contiguously in bf16: row n gets K_in ih-columns
// followed by Hn hh-columns.  Makes the K-chunk staging source a single
// strided pointer (no per-chunk source select in the hot loop).
__global__ void k_pack_wcat(const float* __restrict__ Wih,
                            const float* __restrict__ Whh,
                            uint16_t* __restrict__ out, int K_in) {
    const int KC = K_in + Hn;
    const int total = 4 * Hn * KC;
    int i = blockIdx.x * blockDim.x + threadIdx.x;
    if (i >= total) return;
    int n = i / KC, k = i % KC;
    float v = (k < K_in) ? Wih[n * K_in + k] : Whh[n * Hn + (k - K_in)];
    out[i] = f2bf(v);
}

__global__ void k_bias_sum(const float* __restrict__ a,
                           const float* __restrict__ b,
                           float* __restrict__ o, int n) {
    int i = blockIdx.x * blockDim.x + threadIdx.x;
    if (i < n) o[i] = a[i] + b[i];
}

// ---------------------------------------------------------------------------
// One LSTM timestep for one layer, all-WMMA, async-LDS staged weights.
//
// Gate matrix G = X_t (Bn x K_in) * Wih^T + H_prev (Bn x Hn) * Whh^T, shape
// (Bn x 4*Hn).  All 8 waves of a block share one j-tile; the 4-gate weight
// panel for each 64-wide K-round (64 Wcat rows x 64 k, bf16 = 8 KB) is staged
// into LDS once per block with global_load_async_to_lds_b128, double-buffered
// (one s_wait_asynccnt + barrier per 8 WMMAs), and read with ds_load_b128.
// A fragments are per-wave global_load_b128.
//
// Fragment layouts (CDNA5 ISA 7.12.2, wave32):
//   A 16x32 bf16 : lane L -> row m0 + (L&15); elems 0..7  = K kc+kh..kc+kh+7,
//                  elems 8..15 = K kc+kh+16.. ; kh = (L<16)?0:8
//   B 32x16 bf16 : lane L -> col j  (L&15);   elems 0..7  = K kc+kb..kc+kb+7,
//                  elems 8..15 = K kc+kb+16.. ; kb = (L<16)?0:8   (Wcat rows are k-contiguous)
//   C/D 16x16 f32: VGPR r -> m = m0 + r + ((L<16)?0:8), n = L&15
// ---------------------------------------------------------------------------
#define ROW_PITCH 72            // 64 staged elems + 8 pad (16B-aligned rows)
#define BUF_ELEMS (64 * ROW_PITCH)

__global__ __launch_bounds__(256)
void k_lstm_step_wmma(const uint16_t* __restrict__ Xbf, int x_stride, int K_in,
                      const uint16_t* __restrict__ Wcat,  // (4H, K_in+H) bf16
                      const float* __restrict__ bsum,     // (4H) b_ih+b_hh
                      const uint16_t* __restrict__ Hprev, // (B, H) bf16
                      float* __restrict__ C,              // (B, H) f32 in/out
                      uint16_t* __restrict__ Hout,        // (B, H) bf16
                      float* __restrict__ Hf32)           // (B, H) f32 or null
{
    __shared__ uint16_t smem[2][BUF_ELEMS];

    const int lane = threadIdx.x & 31;
    const int wave = threadIdx.x >> 5;
    const int jt   = blockIdx.x >> 1;                 // 32 j-tiles over H=512
    const int m0   = ((blockIdx.x & 1) * 8 + wave) * 16;  // 16 m-tiles over B
    const int j0   = jt * 16;
    const int ncol = lane & 15;
    const int khA  = (lane < 16) ? 0 : 8;
    const int khB  = (lane < 16) ? 0 : 8;
    const int mrow = m0 + ncol;
    const int KC   = K_in + Hn;

    // staging role: thread i moves 32B (16 elems) of Wcat row (i>>2),
    // element offset (i&3)*16 within the 64-wide round
    const int srow = threadIdx.x >> 2;                // 0..63 = gate*16 + col
    const int sseg = threadIdx.x & 3;                 // 0..3 (16 elems each)
    const int swn  = (srow >> 4) * Hn + j0 + (srow & 15);   // Wcat row index
    const uint16_t* stage_src = Wcat + (size_t)swn * KC + sseg * 16;
    uint16_t* const stage_dst = &smem[0][0] + srow * ROW_PITCH + sseg * 16;

    const int rih  = K_in >> 6;                       // 64-wide rounds (ih)
    const int rtot = rih + (Hn >> 6);                 // + recurrent rounds

    auto stage = [&](int r) {
        const uint16_t* src = stage_src + (r << 6);
        uint16_t* dst = stage_dst + (r & 1) * BUF_ELEMS;
#ifdef HAVE_ASYNC_LDS
        __builtin_amdgcn_global_load_async_to_lds_b128(
            (gas1_v4i)src, (as3_v4i)dst, 0, 0);
        __builtin_amdgcn_global_load_async_to_lds_b128(
            (gas1_v4i)(src + 8), (as3_v4i)(dst + 8), 0, 0);
#else
        *(float4*)dst       = *(const float4*)src;
        *(float4*)(dst + 8) = *(const float4*)(src + 8);
#endif
    };
    auto stage_wait = [&]() {
#ifdef HAVE_ASYNC_LDS
        asm volatile("s_wait_asynccnt 0" ::: "memory");
#endif
        __syncthreads();
    };

    v8f acc[4];
    #pragma unroll
    for (int g = 0; g < 4; ++g) acc[g] = (v8f){0.f,0.f,0.f,0.f,0.f,0.f,0.f,0.f};

    const uint16_t* arow_x = Xbf   + (size_t)mrow * x_stride;
    const uint16_t* arow_h = Hprev + (size_t)mrow * Hn;

    // one 64-wide round = 2 sub-chunks x 4 gates = 8 WMMAs from LDS buffer
    auto compute_round = [&](const uint16_t* arow64, int buf) {
        const uint16_t* bb = &smem[buf][0] + khB;
        #pragma unroll
        for (int u = 0; u < 2; ++u) {
            FragU a;
            const uint16_t* ar = arow64 + (u << 5);
            a.f[0] = *(const float4*)(ar + khA);
            a.f[1] = *(const float4*)(ar + khA + 16);
            #pragma unroll
            for (int g = 0; g < 4; ++g) {
                FragU b;
                const uint16_t* p = bb + (g * 16 + ncol) * ROW_PITCH + (u << 5);
                b.f[0] = *(const float4*)(p);
                b.f[1] = *(const float4*)(p + 16);
                acc[g] = __builtin_amdgcn_wmma_f32_16x16x32_bf16(
                    false, a.v, false, b.v, (short)0, acc[g], false, false);
            }
        }
    };

    stage(0);
    stage_wait();

    int r = 0;
    for (; r < rih; ++r) {                            // A from X_t
        if (r + 1 < rtot) stage(r + 1);
        compute_round(arow_x + (r << 6), r & 1);
        stage_wait();
    }
    for (; r < rtot; ++r) {                           // A from H_prev
        if (r + 1 < rtot) stage(r + 1);
        const int rh = r - rih;
        if (r + 1 < rtot) __builtin_prefetch(arow_h + ((rh + 1) << 6) + khA, 0, 3);
        compute_round(arow_h + (rh << 6), r & 1);
        stage_wait();
    }

    // ---- LSTM pointwise (gate order: i, f, g, o) ---------------------------
    const int j     = j0 + ncol;
    const int mbase = m0 + ((lane < 16) ? 0 : 8);
    const float bi = bsum[j];
    const float bf = bsum[Hn + j];
    const float bg = bsum[2 * Hn + j];
    const float bo = bsum[3 * Hn + j];

    #pragma unroll
    for (int rr = 0; rr < 8; ++rr) {
        const int m = mbase + rr;
        const size_t idx = (size_t)m * Hn + j;
        const float gi = acc[0][rr] + bi;
        const float gf = acc[1][rr] + bf;
        const float gg = acc[2][rr] + bg;
        const float go = acc[3][rr] + bo;
        const float cn = sigf(gf) * C[idx] + sigf(gi) * tanhf(gg);
        const float hn = sigf(go) * tanhf(cn);
        C[idx]    = cn;
        Hout[idx] = f2bf(hn);
        if (Hf32) Hf32[idx] = hn;
    }
}

// ---------------------------------------------------------------------------
// MLP head (tiny: ~67 MFLOP total, plain fp32 VALU is fine)
// ---------------------------------------------------------------------------
__global__ void k_fc1_relu(const float* __restrict__ h,      // (B, H)
                           const float* __restrict__ W,      // (FC, H)
                           const float* __restrict__ bias,
                           float* __restrict__ z) {          // (B, FC)
    int i = blockIdx.x * blockDim.x + threadIdx.x;
    if (i >= Bn * FCn) return;
    int b = i / FCn, jj = i % FCn;
    const float* hr = h + (size_t)b * Hn;
    const float* wr = W + (size_t)jj * Hn;
    float s = bias[jj];
    for (int k = 0; k < Hn; ++k) s += hr[k] * wr[k];
    z[i] = s > 0.f ? s : 0.f;
}

__global__ void k_fc2(const float* __restrict__ z,           // (B, FC)
                      const float* __restrict__ W,           // (OUT, FC)
                      const float* __restrict__ bias,
                      float* __restrict__ out) {             // (B, OUT)
    int i = blockIdx.x * blockDim.x + threadIdx.x;
    if (i >= Bn * OUTn) return;
    int b = i / OUTn, o = i % OUTn;
    const float* zr = z + (size_t)b * FCn;
    const float* wr = W + (size_t)o * FCn;
    float s = bias[o];
    for (int k = 0; k < FCn; ++k) s += zr[k] * wr[k];
    out[i] = s;
}

// ---------------------------------------------------------------------------
// Host driver
// ---------------------------------------------------------------------------
extern "C" void kernel_launch(void* const* d_in, const int* in_sizes, int n_in,
                              void* d_out, int out_size, void* d_ws, size_t ws_size,
                              hipStream_t stream) {
    const float* x     = (const float*)d_in[0];
    const float* W_ih0 = (const float*)d_in[1];
    const float* W_hh0 = (const float*)d_in[2];
    const float* b_ih0 = (const float*)d_in[3];
    const float* b_hh0 = (const float*)d_in[4];
    const float* W_ih1 = (const float*)d_in[5];
    const float* W_hh1 = (const float*)d_in[6];
    const float* b_ih1 = (const float*)d_in[7];
    const float* b_hh1 = (const float*)d_in[8];
    const float* W_fc1 = (const float*)d_in[9];
    const float* b_fc1 = (const float*)d_in[10];
    const float* W_fc2 = (const float*)d_in[11];
    const float* b_fc2 = (const float*)d_in[12];
    float* out = (float*)d_out;

    // --- workspace carve-up (256B aligned), ~27 MB total --------------------
    char*  base = (char*)d_ws;
    size_t off  = 0;
    auto carve = [&](size_t bytes) -> char* {
        char* p = base + off;
        off = (off + bytes + 255) & ~(size_t)255;
        return p;
    };
    uint16_t* x_bf   = (uint16_t*)carve((size_t)Bn * Tn * Dn * 2);
    uint16_t* wcat0  = (uint16_t*)carve((size_t)4 * Hn * (Dn + Hn) * 2);
    uint16_t* wcat1  = (uint16_t*)carve((size_t)4 * Hn * (Hn + Hn) * 2);
    float*    bsum0  = (float*)carve((size_t)4 * Hn * 4);
    float*    bsum1  = (float*)carve((size_t)4 * Hn * 4);
    uint16_t* h0buf[2] = { (uint16_t*)carve((size_t)Bn * Hn * 2),
                           (uint16_t*)carve((size_t)Bn * Hn * 2) };
    uint16_t* h1buf[2] = { (uint16_t*)carve((size_t)Bn * Hn * 2),
                           (uint16_t*)carve((size_t)Bn * Hn * 2) };
    float* c0   = (float*)carve((size_t)Bn * Hn * 4);
    float* c1   = (float*)carve((size_t)Bn * Hn * 4);
    float* h1f  = (float*)carve((size_t)Bn * Hn * 4);
    float* zbuf = (float*)carve((size_t)Bn * FCn * 4);
    (void)ws_size; (void)in_sizes; (void)n_in; (void)out_size;

    // --- one-time packing / conversion --------------------------------------
    {
        int n = Bn * Tn * Dn;
        k_f32_to_bf16<<<(n + 255) / 256, 256, 0, stream>>>(x, x_bf, n);
        int n0 = 4 * Hn * (Dn + Hn);
        k_pack_wcat<<<(n0 + 255) / 256, 256, 0, stream>>>(W_ih0, W_hh0, wcat0, Dn);
        int n1 = 4 * Hn * (Hn + Hn);
        k_pack_wcat<<<(n1 + 255) / 256, 256, 0, stream>>>(W_ih1, W_hh1, wcat1, Hn);
        k_bias_sum<<<(4 * Hn + 255) / 256, 256, 0, stream>>>(b_ih0, b_hh0, bsum0, 4 * Hn);
        k_bias_sum<<<(4 * Hn + 255) / 256, 256, 0, stream>>>(b_ih1, b_hh1, bsum1, 4 * Hn);
    }

    // --- zero initial states ------------------------------------------------
    (void)hipMemsetAsync(h0buf[0], 0, (size_t)Bn * Hn * 2, stream);
    (void)hipMemsetAsync(h0buf[1], 0, (size_t)Bn * Hn * 2, stream);
    (void)hipMemsetAsync(h1buf[0], 0, (size_t)Bn * Hn * 2, stream);
    (void)hipMemsetAsync(h1buf[1], 0, (size_t)Bn * Hn * 2, stream);
    (void)hipMemsetAsync(c0, 0, (size_t)Bn * Hn * 4, stream);
    (void)hipMemsetAsync(c1, 0, (size_t)Bn * Hn * 4, stream);

    // --- sequential scan: 2 layers x T steps --------------------------------
    for (int t = 0; t < Tn; ++t) {
        const int cur = t & 1, nxt = (t + 1) & 1;
        // layer 0: input x[:, t, :], K=64
        k_lstm_step_wmma<<<64, 256, 0, stream>>>(
            x_bf + (size_t)t * Dn, Tn * Dn, Dn,
            wcat0, bsum0,
            h0buf[cur], c0, h0buf[nxt], (float*)nullptr);
        // layer 1: input h0[t], K=512
        k_lstm_step_wmma<<<64, 256, 0, stream>>>(
            h0buf[nxt], Hn, Hn,
            wcat1, bsum1,
            h1buf[cur], c1, h1buf[nxt], h1f);
    }

    // --- MLP head on last hidden state --------------------------------------
    k_fc1_relu<<<(Bn * FCn + 255) / 256, 256, 0, stream>>>(h1f, W_fc1, b_fc1, zbuf);
    k_fc2<<<(Bn * OUTn + 255) / 256, 256, 0, stream>>>(zbuf, W_fc2, b_fc2, out);
}
